// Model_70454643523969
// MI455X (gfx1250) — compile-verified, matrix operand order checked by reading, not run on hardware
//
#include <hip/hip_runtime.h>
#include <math.h>

// ---------------- problem constants ----------------
#define Bn   64
#define Tn   8
#define Sn   512
#define Hn   512
#define Ln   32
#define TPB  256   // 8 wave32

// converted-weight offsets (elements) inside d_ws (bf16)
#define OFF_WATT  0u          // (512,1024)
#define OFF_WFA   524288u     // (512,1536)
#define OFF_WFIS  1310720u    // (512,1024)
#define OFF_WIU   1835008u    // (1536,512)
#define OFF_WID   2621440u    // (1536,512)

typedef __attribute__((ext_vector_type(16))) __bf16 v16bf;
typedef __attribute__((ext_vector_type(8)))  __bf16 v8bf;
typedef __attribute__((ext_vector_type(8)))  float  v8f;
typedef __attribute__((ext_vector_type(4)))  float  f4;

// ------------- f32 -> bf16 weight conversion -------------
__global__ void cvt_bf16_kernel(const float* __restrict__ src,
                                __bf16* __restrict__ dst, int n) {
    int i = blockIdx.x * TPB + threadIdx.x;
    if (i < n) dst[i] = (__bf16)src[i];
}

// ------------- GEMM: y(N) = x(K) @ W(N,K)^T + bias -------------
// K loop outermost: A fragment built once per k-step, reused across the
// wave's NTPW output tiles (independent WMMA chains hide D->C hazards).
// A layout (16-bit 16x32): lane0 e0..7 = K0..7, e8..15 = K16..23;
//                          lane16 e0..7 = K8..15, e8..15 = K24..31.
template <int KT, int NTPW>
__device__ __forceinline__ void gemm_row(const float* __restrict__ x,
                                         const __bf16* __restrict__ W,
                                         const float* __restrict__ bias,
                                         float* __restrict__ y) {
    const int K    = KT * 32;
    const int tid  = threadIdx.x;
    const int lane = tid & 31;
    const int wv   = tid >> 5;
    const int s0   = (lane & 16) ? 8 : 0;
    const float am = ((lane == 0) || (lane == 16)) ? 1.0f : 0.0f;
    const int col  = lane & 15;

    v8f acc[NTPW];
    {
        v8f z = {};
#pragma unroll
        for (int j = 0; j < NTPW; ++j) acc[j] = z;
    }
    for (int kt = 0; kt < KT; ++kt) {
        const int k0 = kt * 32;
        // A fragment (row 0 = x, everything else 0) -- built once per k-step
        f4 xa = *(const f4*)(x + k0 + s0);
        f4 xb = *(const f4*)(x + k0 + s0 + 4);
        f4 xc = *(const f4*)(x + k0 + s0 + 16);
        f4 xd = *(const f4*)(x + k0 + s0 + 20);
        v16bf a;
#pragma unroll
        for (int e = 0; e < 4; ++e) {
            a[e]      = (__bf16)(xa[e] * am);
            a[e + 4]  = (__bf16)(xb[e] * am);
            a[e + 8]  = (__bf16)(xc[e] * am);
            a[e + 12] = (__bf16)(xd[e] * am);
        }
#pragma unroll
        for (int j = 0; j < NTPW; ++j) {
            const int nbase = (wv + 8 * j) << 4;
            const __bf16* wrow = W + (size_t)(nbase + col) * (size_t)K + s0 + k0;
            v8bf blo = *(const v8bf*)wrow;
            v8bf bhi = *(const v8bf*)(wrow + 16);
            v16bf b;
#pragma unroll
            for (int e = 0; e < 8; ++e) { b[e] = blo[e]; b[e + 8] = bhi[e]; }
            acc[j] = __builtin_amdgcn_wmma_f32_16x16x32_bf16(
                false, a, false, b, (short)0, acc[j], false, false);
        }
    }
#pragma unroll
    for (int j = 0; j < NTPW; ++j) {
        const int nbase = (wv + 8 * j) << 4;
        if (lane < 16) y[nbase + lane] = acc[j][0] + bias[nbase + lane];
    }
}

// ------------- online-softmax attention over one source -------------
// V streamed exactly once per use via async global->LDS DMA (ASYNCcnt),
// 16-row double-buffered tiles; per-thread acc owns columns tid, tid+256.
__device__ __forceinline__ void attn_source(const float* __restrict__ Vg,
                                            const int nrows,
                                            const float* __restrict__ kv,
                                            float* __restrict__ osec,
                                            float* __restrict__ sV,      // 2*16*512
                                            float* __restrict__ s_sc,    // 16
                                            float* __restrict__ s_w,     // 16
                                            float* __restrict__ s_soft)  // 4
{
    const int tid = threadIdx.x;
    if (tid == 0) { s_soft[0] = -1e30f; s_soft[1] = 0.0f; }
    float acc0 = 0.0f, acc1 = 0.0f;
    const int c0 = tid, c1 = tid + 256;
    const int nch = (nrows + 15) >> 4;
    // wave-relative LDS byte offset of sV (flat-LDS aperture: offset = addr[31:0])
    const unsigned ldsbase = (unsigned)(unsigned long long)(uintptr_t)sV;

    // issue one 16-row chunk (32 KB) as 8 x b128 async copies per thread
    auto issue = [&](int c, int buf) {
        const int rbase = c << 4;
#pragma unroll
        for (int q = 0; q < 8; ++q) {
            int f = q * TPB + tid, row = f >> 7, cc = f & 127;
            int gr = rbase + row; if (gr >= nrows) gr = nrows - 1;
            unsigned ldsoff = ldsbase + (unsigned)(buf * 32768 + row * 2048 + cc * 16);
            unsigned goff   = (unsigned)(gr * 2048 + cc * 16);
            asm volatile("global_load_async_to_lds_b128 %0, %1, %2 th:TH_LOAD_NT"
                         :: "v"(ldsoff), "v"(goff), "s"(Vg) : "memory");
        }
    };

    // prologue: fill buffer 0
    issue(0, 0);
    asm volatile("s_wait_asynccnt 0x0" ::: "memory");
    __syncthreads();

    for (int c = 0; c < nch; ++c) {
        const int buf = c & 1;
        const bool more = (c + 1 < nch);
        if (more) issue(c + 1, buf ^ 1);   // DMA overlaps compute on `buf`

        // scores: 16 lanes per row, 32 elems each, shfl-reduce
        {
            const float* vr = sV + buf * 8192 + (tid >> 4) * Hn + (tid & 15) * 32;
            const float* kp = kv + (tid & 15) * 32;
            float p = 0.0f;
#pragma unroll
            for (int e = 0; e < 32; ++e) p += vr[e] * kp[e];
            p += __shfl_xor(p, 1); p += __shfl_xor(p, 2);
            p += __shfl_xor(p, 4); p += __shfl_xor(p, 8);
            if ((tid & 15) == 0) s_sc[tid >> 4] = p;
        }
        __syncthreads();
        if (tid < 16) {  // online-softmax bookkeeping
            float sc = ((c << 4) + tid < nrows) ? s_sc[tid] : -1e30f;
            float cm = sc;
            cm = fmaxf(cm, __shfl_xor(cm, 1)); cm = fmaxf(cm, __shfl_xor(cm, 2));
            cm = fmaxf(cm, __shfl_xor(cm, 4)); cm = fmaxf(cm, __shfl_xor(cm, 8));
            float mold = s_soft[0];
            float nm = fmaxf(mold, cm);
            float w = __expf(sc - nm);
            float ws = w;
            ws += __shfl_xor(ws, 1); ws += __shfl_xor(ws, 2);
            ws += __shfl_xor(ws, 4); ws += __shfl_xor(ws, 8);
            s_w[tid] = w;
            if (tid == 0) {
                float rs = __expf(mold - nm);
                s_soft[2] = rs;
                s_soft[0] = nm;
                s_soft[1] = s_soft[1] * rs + ws;
            }
        }
        __syncthreads();
        {
            const float scale = s_soft[2];
            acc0 *= scale; acc1 *= scale;
            const float* vb = sV + buf * 8192;
#pragma unroll
            for (int r = 0; r < 16; ++r) {
                float w = s_w[r];
                acc0 += w * vb[r * Hn + c0];
                acc1 += w * vb[r * Hn + c1];
            }
        }
        if (more) asm volatile("s_wait_asynccnt 0x0" ::: "memory");
        __syncthreads();   // publish buf^1 (async) + everyone done reading buf
    }
    const float linv = 1.0f / s_soft[1];
    osec[c0] = acc0 * linv;
    osec[c1] = acc1 * linv;
    __syncthreads();
}

// ------------- GRU cell with h == 0  (out = (1-z)*n) -------------
__device__ __forceinline__ void gru_zero_h(const float* __restrict__ gi,
                                           const float* __restrict__ bh,
                                           float* __restrict__ o) {
    const int tid = threadIdx.x;
#pragma unroll
    for (int j = tid; j < 512; j += TPB) {
        float r = 1.0f / (1.0f + __expf(-(gi[j] + bh[j])));
        float z = 1.0f / (1.0f + __expf(-(gi[512 + j] + bh[512 + j])));
        float n = tanhf(gi[1024 + j] + r * bh[1024 + j]);
        o[j] = (1.0f - z) * n;
    }
}

// ------------- main persistent kernel: 1 block per batch -------------
__global__ __launch_bounds__(TPB, 1)
void editsql_kernel(const float* __restrict__ feat,   // (B,T,S,H)
                    const float* __restrict__ dec,    // (B,L,H)
                    const float* __restrict__ b_att,
                    const float* __restrict__ b_fa,
                    const float* __restrict__ b_fis,
                    const float* __restrict__ bi_u,
                    const float* __restrict__ bh_u,
                    const float* __restrict__ bi_d,
                    const float* __restrict__ bh_d,
                    const __bf16* __restrict__ Wb,    // converted weights
                    float* __restrict__ out)          // (B,L,H)
{
    __shared__ float s_u[512];
    __shared__ float s_d[512];
    __shared__ float s_uall[Tn][512];
    __shared__ float s_k[512];
    __shared__ float s_x[1536];     // GEMM input / attention output concat
    __shared__ float s_fuse[512];
    __shared__ float s_gi[1536];
    __shared__ float s_V[2 * 16 * 512];  // 64 KB attention tiles
    __shared__ float s_sc[16];
    __shared__ float s_w[16];
    __shared__ float s_soft[4];

    const int tid = threadIdx.x;
    const int b   = blockIdx.x;
    const __bf16* Watt = Wb + OFF_WATT;
    const __bf16* Wfa  = Wb + OFF_WFA;
    const __bf16* Wfis = Wb + OFF_WFIS;
    const __bf16* Wiu  = Wb + OFF_WIU;
    const __bf16* Wid  = Wb + OFF_WID;

    for (int j = tid; j < 512; j += TPB) s_d[j] = 0.0f;
    __syncthreads();

    for (int i = 0; i < Ln; ++i) {
        for (int j = tid; j < 512; j += TPB) s_u[j] = 0.0f;
        __syncthreads();

        for (int t = 0; t < Tn; ++t) {
            // k = [u, d] @ W_att^T + b_att
            for (int j = tid; j < 512; j += TPB) { s_x[j] = s_u[j]; s_x[512 + j] = s_d[j]; }
            __syncthreads();
            gemm_row<32, 4>(s_x, Watt, b_att, s_k);   // K=1024, N=512
            __syncthreads();

            // three attention sums (db / utter / sql), results -> s_x[0..1535]
            const float* vb = feat + ((size_t)(b * Tn + t)) * Sn * Hn;
            attn_source(vb,             128, s_k, s_x,        s_V, s_sc, s_w, s_soft);
            attn_source(vb + 129 * Hn,  191, s_k, s_x + 512,  s_V, s_sc, s_w, s_soft);
            attn_source(vb + 321 * Hn,  191, s_k, s_x + 1024, s_V, s_sc, s_w, s_soft);

            // fuse = att @ W_fa^T + b_fa ; gi = fuse @ Wi_u^T + bi_u ; GRU
            gemm_row<48, 4>(s_x, Wfa, b_fa, s_fuse);  // K=1536, N=512
            __syncthreads();
            gemm_row<16, 12>(s_fuse, Wiu, bi_u, s_gi); // K=512, N=1536
            __syncthreads();
            gru_zero_h(s_gi, bh_u, s_u);
            __syncthreads();
            for (int j = tid; j < 512; j += TPB) s_uall[t][j] = s_u[j];
            __syncthreads();
        }

        // u_sum = attn(u_last, u_all[0..7])
        {
            const int lane = tid & 31, wv = tid >> 5;
            float p = 0.0f;
            for (int e = lane; e < 512; e += 32) p += s_u[e] * s_uall[wv][e];
            p += __shfl_xor(p, 1); p += __shfl_xor(p, 2); p += __shfl_xor(p, 4);
            p += __shfl_xor(p, 8); p += __shfl_xor(p, 16);
            if (lane == 0) s_sc[wv] = p;
            __syncthreads();
            if (tid == 0) {
                float mx = -1e30f;
                for (int t = 0; t < Tn; ++t) mx = fmaxf(mx, s_sc[t]);
                float sm = 0.0f;
                for (int t = 0; t < Tn; ++t) { float w = __expf(s_sc[t] - mx); s_w[t] = w; sm += w; }
                s_soft[1] = sm;
            }
            __syncthreads();
            const float linv = 1.0f / s_soft[1];
            for (int j = tid; j < 512; j += TPB) {
                float a = 0.0f;
#pragma unroll
                for (int t = 0; t < Tn; ++t) a += s_w[t] * s_uall[t][j];
                s_x[j] = a * linv;
            }
        }
        // concat x_i
        {
            const float* xi = dec + ((size_t)b * Ln + i) * Hn;
            for (int j = tid; j < 512; j += TPB) s_x[512 + j] = xi[j];
        }
        __syncthreads();
        gemm_row<32, 4>(s_x, Wfis, b_fis, s_fuse);  // K=1024, N=512
        __syncthreads();
        gemm_row<16, 12>(s_fuse, Wid, bi_d, s_gi);  // K=512, N=1536
        __syncthreads();
        gru_zero_h(s_gi, bh_d, s_d);
        __syncthreads();
        for (int j = tid; j < 512; j += TPB)
            out[((size_t)b * Ln + i) * Hn + j] = s_d[j];
        __syncthreads();
    }
}

// ------------------------- launcher -------------------------
extern "C" void kernel_launch(void* const* d_in, const int* in_sizes, int n_in,
                              void* d_out, int out_size, void* d_ws, size_t ws_size,
                              hipStream_t stream) {
    (void)in_sizes; (void)n_in; (void)out_size; (void)ws_size;
    const float* feat  = (const float*)d_in[0];
    const float* dec   = (const float*)d_in[1];
    const float* W_att = (const float*)d_in[2];
    const float* b_att = (const float*)d_in[3];
    const float* W_fa  = (const float*)d_in[4];
    const float* b_fa  = (const float*)d_in[5];
    const float* W_fis = (const float*)d_in[6];
    const float* b_fis = (const float*)d_in[7];
    const float* Wi_u  = (const float*)d_in[8];
    // d_in[9]  = Wh_u  (multiplied by h==0 -> unused)
    const float* bi_u  = (const float*)d_in[10];
    const float* bh_u  = (const float*)d_in[11];
    const float* Wi_d  = (const float*)d_in[12];
    // d_in[13] = Wh_d  (unused)
    const float* bi_d  = (const float*)d_in[14];
    const float* bh_d  = (const float*)d_in[15];

    __bf16* wb = (__bf16*)d_ws;   // needs ~6.8 MB of workspace

    cvt_bf16_kernel<<<(524288 + TPB - 1) / TPB, TPB, 0, stream>>>(W_att, wb + OFF_WATT, 524288);
    cvt_bf16_kernel<<<(786432 + TPB - 1) / TPB, TPB, 0, stream>>>(W_fa,  wb + OFF_WFA,  786432);
    cvt_bf16_kernel<<<(524288 + TPB - 1) / TPB, TPB, 0, stream>>>(W_fis, wb + OFF_WFIS, 524288);
    cvt_bf16_kernel<<<(786432 + TPB - 1) / TPB, TPB, 0, stream>>>(Wi_u,  wb + OFF_WIU,  786432);
    cvt_bf16_kernel<<<(786432 + TPB - 1) / TPB, TPB, 0, stream>>>(Wi_d,  wb + OFF_WID,  786432);

    editsql_kernel<<<Bn, TPB, 0, stream>>>(feat, dec, b_att, b_fa, b_fis,
                                           bi_u, bh_u, bi_d, bh_d,
                                           wb, (float*)d_out);
}
